// VectorQuantizerEMA_17592186045166
// MI455X (gfx1250) — matrix operand order repaired and checked by reading, not compiled, not run on hardware
//
#include <hip/hip_runtime.h>
#include <hip/hip_bf16.h>

typedef __attribute__((ext_vector_type(16))) _Float16 v16h;
typedef __attribute__((ext_vector_type(4)))  _Float16 v4h;
typedef __attribute__((ext_vector_type(8)))  float    v8f;

#define NUM_VAR 64
#define BATCH   4096
#define EMB_DIM 64
#define NUM_EMB 512
#define ROWS_PER_WAVE 32                       // two 16-row strips per wave
#define ROWS_PER_WG   256                      // 8 waves * 32 rows
#define CHUNKS (BATCH / ROWS_PER_WG)           // 16
#define WGS    (NUM_VAR * CHUNKS)              // 1024

// Kernel 1: per (v, 256-row chunk). f16 WMMA GEMM -> argmin -> fp32 gather/output/loss partial.
// Each wave owns TWO 16-row strips so each B tile read from LDS feeds 4 WMMAs
// (halves LDS bytes per WMMA; LDS port rate is the limiter otherwise).
__global__ __launch_bounds__(256) void vq_wmma_kernel(
    const float* __restrict__ x,      // [V, B, D]
    const float* __restrict__ emb,    // [V, D, K]
    float* __restrict__ out,          // [V, B, D]
    float* __restrict__ wsLoss)       // [WGS] partial loss sums
{
    __shared__ _Float16 ldsB[EMB_DIM * NUM_EMB];  // 64KB: f16 B-tiles, WMMA layout
    __shared__ float    ldsW2[NUM_EMB];           // ||w_k||^2 (fp32)
    __shared__ float    ldsPartial[8];

    const int tid   = threadIdx.x;
    const int lane  = tid & 31;
    const int wave  = tid >> 5;
    const int m     = lane & 15;   // row-in-strip / N-in-tile for this lane
    const int h     = lane >> 4;   // half-wave
    const int wg    = blockIdx.x;
    const int v     = wg >> 4;     // / CHUNKS
    const int chunk = wg & (CHUNKS - 1);
    const int rowBase = chunk * ROWS_PER_WG + wave * ROWS_PER_WAVE;

    const float* embV = emb + (size_t)v * EMB_DIM * NUM_EMB;

    // ---- Stage embeddings[v] into LDS as f16 B-tiles (vectorized: float4 -> 4xf16) ----
    // ldsB index i = t*1024 + s*512 + L*16 + e  holds emb[v, d = s*32 + (e/8)*16 + (L/16)*8 + e%8, code = t*16 + L%16]
    for (int idx = tid; idx < (EMB_DIM * NUM_EMB) / 4; idx += 256) {
        int i = idx * 4;
        int e = i & 15;
        int L = (i >> 4) & 31;
        int s = (i >> 9) & 1;
        int t = i >> 10;
        int d = s * 32 + (e >> 3) * 16 + (L >> 4) * 8 + (e & 7);
        int code = t * 16 + (L & 15);
        const float* ep = embV + (size_t)d * NUM_EMB + code;
        float4 f = make_float4(ep[0], ep[NUM_EMB], ep[2 * NUM_EMB], ep[3 * NUM_EMB]);
        v4h hv;
        hv[0] = (_Float16)f.x; hv[1] = (_Float16)f.y;
        hv[2] = (_Float16)f.z; hv[3] = (_Float16)f.w;
        *(v4h*)(&ldsB[i]) = hv;
    }
    // ---- ||w_k||^2 in fp32 (coalesced over codes; L2-resident re-read) ----
    for (int c0 = tid; c0 < NUM_EMB; c0 += 256) {
        float acc = 0.f;
        #pragma unroll 8
        for (int d = 0; d < EMB_DIM; ++d) {
            float w = embV[(size_t)d * NUM_EMB + c0];
            acc += w * w;
        }
        ldsW2[c0] = acc;
    }
    __syncthreads();

    // ---- Load A strips (2 x 16 rows x 64 D) as 16x32 f16 tiles, documented layout ----
    // lane: row = m, half h; element e -> d = s*32 + (e/8)*16 + h*8 + (e%8)
    v16h a[2][2];
    #pragma unroll
    for (int p = 0; p < 2; ++p) {
        const float* xrow = x + ((size_t)v * BATCH + rowBase + p * 16 + m) * EMB_DIM;
        #pragma unroll
        for (int s = 0; s < 2; ++s) {
            int d0 = s * 32 + h * 8;
            float4 lo0 = *(const float4*)(xrow + d0);
            float4 lo1 = *(const float4*)(xrow + d0 + 4);
            float4 hi0 = *(const float4*)(xrow + d0 + 16);
            float4 hi1 = *(const float4*)(xrow + d0 + 20);
            a[p][s][0]  = (_Float16)lo0.x; a[p][s][1]  = (_Float16)lo0.y;
            a[p][s][2]  = (_Float16)lo0.z; a[p][s][3]  = (_Float16)lo0.w;
            a[p][s][4]  = (_Float16)lo1.x; a[p][s][5]  = (_Float16)lo1.y;
            a[p][s][6]  = (_Float16)lo1.z; a[p][s][7]  = (_Float16)lo1.w;
            a[p][s][8]  = (_Float16)hi0.x; a[p][s][9]  = (_Float16)hi0.y;
            a[p][s][10] = (_Float16)hi0.z; a[p][s][11] = (_Float16)hi0.w;
            a[p][s][12] = (_Float16)hi1.x; a[p][s][13] = (_Float16)hi1.y;
            a[p][s][14] = (_Float16)hi1.z; a[p][s][15] = (_Float16)hi1.w;
        }
    }

    // ---- GEMM sweep over 32 N-tiles; track argmin of (||w||^2 - 2*dot) per strip ----
    float bestVal[2][8];
    int   bestIdx[2][8];
    #pragma unroll
    for (int p = 0; p < 2; ++p)
        #pragma unroll
        for (int r = 0; r < 8; ++r) { bestVal[p][r] = 3.402823466e38f; bestIdx[p][r] = 0; }

    for (int t = 0; t < 32; ++t) {
        v16h b0 = *(const v16h*)(&ldsB[t * 1024 +       lane * 16]);
        v16h b1 = *(const v16h*)(&ldsB[t * 1024 + 512 + lane * 16]);

        v8f c0 = {0.f, 0.f, 0.f, 0.f, 0.f, 0.f, 0.f, 0.f};
        v8f c1 = {0.f, 0.f, 0.f, 0.f, 0.f, 0.f, 0.f, 0.f};
        c0 = __builtin_amdgcn_wmma_f32_16x16x32_f16(false, a[0][0], false, b0, (short)0, c0, false, false);
        c0 = __builtin_amdgcn_wmma_f32_16x16x32_f16(false, a[0][1], false, b1, (short)0, c0, false, false);
        c1 = __builtin_amdgcn_wmma_f32_16x16x32_f16(false, a[1][0], false, b0, (short)0, c1, false, false);
        c1 = __builtin_amdgcn_wmma_f32_16x16x32_f16(false, a[1][1], false, b1, (short)0, c1, false, false);

        float w2n = ldsW2[t * 16 + m];
        int code  = t * 16 + m;
        #pragma unroll
        for (int r = 0; r < 8; ++r) {
            float s0 = w2n - 2.0f * c0[r];
            if (s0 < bestVal[0][r]) { bestVal[0][r] = s0; bestIdx[0][r] = code; }
            float s1 = w2n - 2.0f * c1[r];
            if (s1 < bestVal[1][r]) { bestVal[1][r] = s1; bestIdx[1][r] = code; }
        }
    }

    // ---- Reduce argmin across the 16 lanes of each half-wave (first-min tie rule) ----
    #pragma unroll
    for (int p = 0; p < 2; ++p) {
        #pragma unroll
        for (int r = 0; r < 8; ++r) {
            #pragma unroll
            for (int msk = 1; msk < 16; msk <<= 1) {
                float ov = __shfl_xor(bestVal[p][r], msk, 32);
                int   oi = __shfl_xor(bestIdx[p][r], msk, 32);
                if (ov < bestVal[p][r] || (ov == bestVal[p][r] && oi < bestIdx[p][r])) {
                    bestVal[p][r] = ov; bestIdx[p][r] = oi;
                }
            }
        }
    }

    // ---- Gather fp32 codebook columns, write output, accumulate loss ----
    float lossAcc = 0.f;
    #pragma unroll
    for (int p = 0; p < 2; ++p) {
        #pragma unroll
        for (int rr = 0; rr < 16; ++rr) {
            int r   = rr & 7;
            int src = (rr >> 3) * 16;   // half-wave holding this row's result
            int code = __shfl(bestIdx[p][r], src, 32);
            size_t rowOff = ((size_t)v * BATCH + rowBase + p * 16 + rr) * EMB_DIM;
            const float* xr   = x + rowOff;
            float*       outr = out + rowOff;
            const float* ecol = embV + code;
            float w0 = ecol[(size_t)lane * NUM_EMB];
            float w1 = ecol[(size_t)(lane + 32) * NUM_EMB];
            float x0 = xr[lane];
            float x1 = xr[lane + 32];
            outr[lane]      = w0;   // straight-through output == quantized (fp32 exact)
            outr[lane + 32] = w1;
            float d0 = w0 - x0, d1 = w1 - x1;
            lossAcc += d0 * d0 + d1 * d1;
        }
    }

    // ---- Deterministic per-workgroup loss partial ----
    #pragma unroll
    for (int msk = 1; msk < 32; msk <<= 1)
        lossAcc += __shfl_xor(lossAcc, msk, 32);
    if (lane == 0) ldsPartial[wave] = lossAcc;
    __syncthreads();
    if (tid == 0) {
        float s = 0.f;
        #pragma unroll
        for (int w = 0; w < 8; ++w) s += ldsPartial[w];
        wsLoss[wg] = s;
    }
}

// Kernel 2: fixed-order reduction of WGS partials -> scalar loss (deterministic).
__global__ __launch_bounds__(256) void vq_loss_reduce(
    const float* __restrict__ ws, float* __restrict__ lossOut)
{
    __shared__ float s[256];
    float acc = 0.f;
    for (int i = threadIdx.x; i < WGS; i += 256) acc += ws[i];
    s[threadIdx.x] = acc;
    __syncthreads();
    for (int st = 128; st > 0; st >>= 1) {
        if (threadIdx.x < st) s[threadIdx.x] += s[threadIdx.x + st];
        __syncthreads();
    }
    if (threadIdx.x == 0) {
        const float scale = 0.25f / (float)((size_t)NUM_VAR * BATCH * EMB_DIM);
        lossOut[0] = scale * s[0];
    }
}

extern "C" void kernel_launch(void* const* d_in, const int* in_sizes, int n_in,
                              void* d_out, int out_size, void* d_ws, size_t ws_size,
                              hipStream_t stream) {
    const float* x   = (const float*)d_in[0];   // [V, B, D]
    const float* emb = (const float*)d_in[1];   // [V, D, K]
    float* out    = (float*)d_out;              // [V*B*D] output then [1] loss
    float* wsLoss = (float*)d_ws;               // WGS partial sums

    vq_wmma_kernel<<<WGS, 256, 0, stream>>>(x, emb, out, wsLoss);
    vq_loss_reduce<<<1, 256, 0, stream>>>(
        wsLoss, out + (size_t)NUM_VAR * BATCH * EMB_DIM);
}
